// GNN_8151847927962
// MI455X (gfx1250) — matrix operand order repaired and checked by reading, not compiled, not run on hardware
//
#include <hip/hip_runtime.h>
#include <cstdint>
#include <cstddef>

// ---------------------------------------------------------------------------
// GCN forward for MI455X (gfx1250, wave32).
//  - GEMM in bf16 via v_wmma_f32_16x16x32_bf16 (f32 accumulate)
//  - aggregation via CSR gather (zero per-feature atomics), fused with
//    self-loop term + LayerNorm + ReLU, one wave per node
//  - ht (51 MB bf16) is L2-resident on the 192 MB L2 -> gather runs at L2 BW
// Workspace usage ~112 MB.
// ---------------------------------------------------------------------------

typedef __attribute__((ext_vector_type(16))) __bf16 v16bf;
typedef __attribute__((ext_vector_type(8)))  float  v8f;

#define D_FEAT 512
#define BM 128
#define BN 64
#define BK 32
#define LDP 34   // padded LDS row stride (elements): 17 banks -> conflict-free

// ---------------------------- degree / CSR build ---------------------------

__global__ void gcn_zero_indeg(int* __restrict__ indeg, int n) {
    int i = blockIdx.x * blockDim.x + threadIdx.x;
    if (i < n) indeg[i] = 0;
}

__global__ void gcn_count_indeg(const int* __restrict__ ei, int* __restrict__ indeg,
                                int nedges) {
    int e = blockIdx.x * blockDim.x + threadIdx.x;
    if (e < nedges) atomicAdd(&indeg[ei[nedges + e]], 1);  // dst row of edge_index
}

__global__ void gcn_finish_deg(const int* __restrict__ indeg,
                               float* __restrict__ inv_sqrt,
                               float* __restrict__ inv_deg, int n) {
    int i = blockIdx.x * blockDim.x + threadIdx.x;
    if (i < n) {
        float d = 1.0f + (float)indeg[i];   // GCN degree includes self-loop
        inv_sqrt[i] = rsqrtf(d);
        inv_deg[i]  = 1.0f / d;
    }
}

// Single-block exclusive scan of indeg -> row_ptr[n+1]; cursor = row_ptr copy.
__global__ void gcn_scan(const int* __restrict__ indeg,
                         int* __restrict__ row_ptr,
                         int* __restrict__ cursor, int n) {
    __shared__ int buf[1024];
    __shared__ int carry_s;
    if (threadIdx.x == 0) carry_s = 0;
    __syncthreads();
    for (int base = 0; base < n; base += 1024) {
        int i = base + (int)threadIdx.x;
        int v = (i < n) ? indeg[i] : 0;
        buf[threadIdx.x] = v;
        __syncthreads();
        #pragma unroll
        for (int off = 1; off < 1024; off <<= 1) {
            int t = (threadIdx.x >= (unsigned)off) ? buf[threadIdx.x - off] : 0;
            __syncthreads();
            buf[threadIdx.x] += t;
            __syncthreads();
        }
        int incl = buf[threadIdx.x];
        int excl = incl - v;
        int carry = carry_s;
        if (i < n) { row_ptr[i] = carry + excl; cursor[i] = carry + excl; }
        __syncthreads();
        if (threadIdx.x == 1023) carry_s = carry + incl;  // incl@1023 == chunk total
        __syncthreads();
    }
    if (threadIdx.x == 0) row_ptr[n] = carry_s;
}

__global__ void gcn_fill_csr(const int* __restrict__ ei,
                             const float* __restrict__ inv_sqrt,
                             int* __restrict__ cursor,
                             int* __restrict__ csr_src,
                             float* __restrict__ csr_w, int nedges) {
    int e = blockIdx.x * blockDim.x + threadIdx.x;
    if (e < nedges) {
        int s = ei[e];
        int d = ei[nedges + e];
        int pos = atomicAdd(&cursor[d], 1);
        csr_src[pos] = s;
        csr_w[pos]   = inv_sqrt[s] * inv_sqrt[d];
    }
}

// ------------------------- f32 -> bf16 conversion --------------------------

__global__ void gcn_f32_to_bf16(const float* __restrict__ in,
                                __bf16* __restrict__ out, size_t n) {
    size_t i = ((size_t)blockIdx.x * blockDim.x + threadIdx.x) * 4;
    if (i + 3 < n) {
        float4 v = *(const float4*)(in + i);
        out[i + 0] = (__bf16)v.x;
        out[i + 1] = (__bf16)v.y;
        out[i + 2] = (__bf16)v.z;
        out[i + 3] = (__bf16)v.w;
    } else {
        for (size_t j = i; j < n; ++j) out[j] = (__bf16)in[j];
    }
}

// ------------------------------ WMMA GEMM ----------------------------------
// Ht[m,n] = bf16( sum_k A[m,k] * W[k,n] + bias[n] )
// Block tile 128x64, 8 waves (4 along M x 2 along N), each wave 32x32
// (2x2 tiles of 16x16).

__launch_bounds__(256, 2)
__global__ void gcn_gemm_wmma(const __bf16* __restrict__ A,   // [nrows, 512] bf16
                              const __bf16* __restrict__ Bw,  // [512, 512] bf16 row-major (k,n)
                              const float*  __restrict__ bias,
                              __bf16* __restrict__ Ht,        // [nrows, 512] bf16
                              int nrows) {
    __shared__ __bf16 sA[BM][LDP];   // [m][k]
    __shared__ __bf16 sB[BN][LDP];   // transposed: [n][k]

    const int tid  = threadIdx.x;
    const int lane = tid & 31;
    const int wave = tid >> 5;       // 0..7
    const int wm   = wave & 3;       // wave position along M (4)
    const int wn   = wave >> 2;      // wave position along N (2)
    const int half = lane >> 4;      // lane group 0/1
    const int l16  = lane & 15;

    const int block_m = blockIdx.y * BM;
    const int block_n = blockIdx.x * BN;

    v8f acc[2][2] = {};              // wave computes 32(M) x 32(N)

    for (int k0 = 0; k0 < D_FEAT; k0 += BK) {
        __syncthreads();
        // ---- stage A tile: 128 rows x 32 k (bf16), 16 elements per thread
        {
            int r  = tid >> 1;               // 0..127
            int c  = (tid & 1) * 16;         // 0 or 16
            int gr = block_m + r;
            if (gr >= nrows) gr = nrows - 1; // clamp; garbage rows never stored
            const uint4* src = (const uint4*)(A + (size_t)gr * D_FEAT + k0 + c);
            uint4 d0 = src[0];
            uint4 d1 = src[1];
            unsigned int* dst = (unsigned int*)&sA[r][c];  // 4B aligned
            dst[0] = d0.x; dst[1] = d0.y; dst[2] = d0.z; dst[3] = d0.w;
            dst[4] = d1.x; dst[5] = d1.y; dst[6] = d1.z; dst[7] = d1.w;
        }
        // ---- stage B tile transposed: W[k0..k0+31][block_n..+63] -> sB[n][k]
        {
            int kk = tid >> 3;               // 0..31
            int n0 = (tid & 7) * 8;          // 0..56
            const __bf16* src = Bw + (size_t)(k0 + kk) * D_FEAT + block_n + n0;
            uint4 d = *(const uint4*)src;    // 8 bf16
            union { uint4 u; __bf16 h[8]; } pk; pk.u = d;
            #pragma unroll
            for (int i = 0; i < 8; ++i) sB[n0 + i][kk] = pk.h[i];
        }
        __syncthreads();

        // ---- build fragments per ISA VGPR layouts and issue WMMA
        v16bf afrag[2], bfrag[2];
        #pragma unroll
        for (int i = 0; i < 2; ++i) {
            const __bf16* ap = &sA[wm * 32 + i * 16 + l16][0];
            #pragma unroll
            for (int v = 0; v < 8; ++v) {
                // 16-bit A 16x32: VGPR v holds K = {2v,2v+1} (v<4) or {2v+8,..},
                // +8 for lanes 16..31
                int kb = ((v < 4) ? 2 * v : 2 * v + 8) + half * 8;
                afrag[i][2 * v]     = ap[kb];
                afrag[i][2 * v + 1] = ap[kb + 1];
            }
        }
        #pragma unroll
        for (int j = 0; j < 2; ++j) {
            const __bf16* bp = &sB[wn * 32 + j * 16 + l16][0];
            #pragma unroll
            for (int v = 0; v < 8; ++v) {
                // B 32x16: same K order as A (K = {2v,2v+1} (v<4) or {2v+8,..}, +8 for lanes 16..31)
                int kb = ((v < 4) ? 2 * v : 2 * v + 8) + half * 8;
                bfrag[j][2 * v]     = bp[kb];
                bfrag[j][2 * v + 1] = bp[kb + 1];
            }
        }
        #pragma unroll
        for (int i = 0; i < 2; ++i)
            #pragma unroll
            for (int j = 0; j < 2; ++j)
                acc[i][j] = __builtin_amdgcn_wmma_f32_16x16x32_bf16(
                    false, afrag[i], false, bfrag[j],
                    (short)0, acc[i][j], false, false);
    }

    // ---- epilogue: + bias, write Ht (bf16); fast path for full tiles
    const bool full = (block_m + BM) <= nrows;
    #pragma unroll
    for (int i = 0; i < 2; ++i) {
        #pragma unroll
        for (int j = 0; j < 2; ++j) {
            int gn = block_n + wn * 32 + j * 16 + l16;
            float bv = bias[gn];
            int mb = block_m + wm * 32 + i * 16 + half * 8;
            if (full) {
                #pragma unroll
                for (int r = 0; r < 8; ++r)
                    Ht[(size_t)(mb + r) * D_FEAT + gn] = (__bf16)(acc[i][j][r] + bv);
            } else {
                #pragma unroll
                for (int r = 0; r < 8; ++r)
                    if (mb + r < nrows)
                        Ht[(size_t)(mb + r) * D_FEAT + gn] = (__bf16)(acc[i][j][r] + bv);
            }
        }
    }
}

// -------------- fused CSR aggregate + self-loop + LayerNorm + ReLU ---------
// One wave per node. Lane owns 16 contiguous features (32B bf16 per edge ->
// one coalesced 1KB read per wave per edge, no atomics). f32 accumulation,
// then wave32 shuffle LN.

__launch_bounds__(256)
__global__ void gcn_agg_ln_relu(const __bf16* __restrict__ Ht,
                                const int*   __restrict__ row_ptr,
                                const int*   __restrict__ csr_src,
                                const float* __restrict__ csr_w,
                                const float* __restrict__ inv_deg,
                                const float* __restrict__ gamma,
                                const float* __restrict__ beta,
                                float*  __restrict__ OutF,   // f32 (d_out)
                                __bf16* __restrict__ OutB,   // bf16 (next GEMM input)
                                int nrows) {
    int wid  = (int)((blockIdx.x * (size_t)blockDim.x + threadIdx.x) >> 5);
    int lane = threadIdx.x & 31;
    if (wid >= nrows) return;

    union U16 { uint4 u[2]; __bf16 h[16]; };
    float acc[16];

    // self-loop term: ht[wid] * inv_deg[wid]
    {
        float idg = inv_deg[wid];
        U16 s;
        const uint4* sp = (const uint4*)(Ht + (size_t)wid * D_FEAT + lane * 16);
        s.u[0] = sp[0]; s.u[1] = sp[1];
        #pragma unroll
        for (int j = 0; j < 16; ++j) acc[j] = idg * (float)s.h[j];
    }

    // gather incoming edges (uniform bounds per wave -> scalar loads)
    int e0 = row_ptr[wid];
    int e1 = row_ptr[wid + 1];
    for (int e = e0; e < e1; ++e) {
        int   s = csr_src[e];
        float w = csr_w[e];
        if (e + 1 < e1)
            __builtin_prefetch(Ht + (size_t)csr_src[e + 1] * D_FEAT + lane * 16, 0, 1);
        U16 a;
        const uint4* sp = (const uint4*)(Ht + (size_t)s * D_FEAT + lane * 16);
        a.u[0] = sp[0]; a.u[1] = sp[1];
        #pragma unroll
        for (int j = 0; j < 16; ++j) acc[j] += w * (float)a.h[j];
    }

    // LayerNorm over 512 (16 per lane + wave32 shuffle reduce)
    float s = 0.0f;
    #pragma unroll
    for (int j = 0; j < 16; ++j) s += acc[j];
    #pragma unroll
    for (int off = 16; off > 0; off >>= 1) s += __shfl_xor(s, off, 32);
    float mu = s * (1.0f / (float)D_FEAT);

    float s2 = 0.0f;
    #pragma unroll
    for (int j = 0; j < 16; ++j) { float d = acc[j] - mu; s2 += d * d; }
    #pragma unroll
    for (int off = 16; off > 0; off >>= 1) s2 += __shfl_xor(s2, off, 32);
    float inv = rsqrtf(s2 * (1.0f / (float)D_FEAT) + 1e-5f);

    const float4* g4 = (const float4*)(gamma) + lane * 4;
    const float4* b4 = (const float4*)(beta)  + lane * 4;
    float4* outf = (float4*)(OutF + (size_t)wid * D_FEAT + lane * 16);
    U16 ob;
    #pragma unroll
    for (int q = 0; q < 4; ++q) {
        float4 g = g4[q], b = b4[q], o;
        o.x = fmaxf(g.x * (acc[q * 4 + 0] - mu) * inv + b.x, 0.0f);
        o.y = fmaxf(g.y * (acc[q * 4 + 1] - mu) * inv + b.y, 0.0f);
        o.z = fmaxf(g.z * (acc[q * 4 + 2] - mu) * inv + b.z, 0.0f);
        o.w = fmaxf(g.w * (acc[q * 4 + 3] - mu) * inv + b.w, 0.0f);
        outf[q] = o;
        ob.h[q * 4 + 0] = (__bf16)o.x;
        ob.h[q * 4 + 1] = (__bf16)o.y;
        ob.h[q * 4 + 2] = (__bf16)o.z;
        ob.h[q * 4 + 3] = (__bf16)o.w;
    }
    uint4* outb = (uint4*)(OutB + (size_t)wid * D_FEAT + lane * 16);
    outb[0] = ob.u[0];
    outb[1] = ob.u[1];
}

// ------------------------------ launcher -----------------------------------

static inline char* ws_take(char*& p, size_t bytes) {
    char* r = p;
    p += (bytes + 255) & ~(size_t)255;
    return r;
}

extern "C" void kernel_launch(void* const* d_in, const int* in_sizes, int n_in,
                              void* d_out, int out_size, void* d_ws, size_t ws_size,
                              hipStream_t stream) {
    const float* x      = (const float*)d_in[0];   // [N, 512]
    const int*   ei     = (const int*)  d_in[1];   // [2, E]
    const float* Ws     = (const float*)d_in[2];   // [3, 512, 512]
    const float* bs     = (const float*)d_in[3];   // [3, 512]
    const float* gammas = (const float*)d_in[4];   // [3, 512]
    const float* betas  = (const float*)d_in[5];   // [3, 512]
    float* out = (float*)d_out;                    // [N, 512]

    const int N = in_sizes[0] / D_FEAT;
    const int E = in_sizes[1] / 2;
    const int L = in_sizes[2] / (D_FEAT * D_FEAT);

    // workspace carve (~112 MB)
    char* p = (char*)d_ws;
    int*    indeg    = (int*)  ws_take(p, (size_t)N * 4);
    float*  inv_sqrt = (float*)ws_take(p, (size_t)N * 4);
    float*  inv_deg  = (float*)ws_take(p, (size_t)N * 4);
    int*    row_ptr  = (int*)  ws_take(p, (size_t)(N + 1) * 4);
    int*    cursor   = (int*)  ws_take(p, (size_t)N * 4);
    int*    csr_src  = (int*)  ws_take(p, (size_t)E * 4);
    float*  csr_w    = (float*)ws_take(p, (size_t)E * 4);
    __bf16* Wb       = (__bf16*)ws_take(p, (size_t)D_FEAT * D_FEAT * 2);
    __bf16* hB       = (__bf16*)ws_take(p, (size_t)N * D_FEAT * 2);
    __bf16* htB      = (__bf16*)ws_take(p, (size_t)N * D_FEAT * 2);
    (void)ws_size; (void)n_in; (void)out_size;

    // ---- CSR build (once per call; in-row edge order not deterministic) ----
    gcn_zero_indeg<<<(N + 255) / 256, 256, 0, stream>>>(indeg, N);
    gcn_count_indeg<<<(E + 255) / 256, 256, 0, stream>>>(ei, indeg, E);
    gcn_finish_deg<<<(N + 255) / 256, 256, 0, stream>>>(indeg, inv_sqrt, inv_deg, N);
    gcn_scan<<<1, 1024, 0, stream>>>(indeg, row_ptr, cursor, N);
    gcn_fill_csr<<<(E + 255) / 256, 256, 0, stream>>>(ei, inv_sqrt, cursor,
                                                      csr_src, csr_w, E);

    // ---- h0 = bf16(x) ----
    {
        size_t n = (size_t)N * D_FEAT;
        gcn_f32_to_bf16<<<(unsigned)((n / 4 + 255) / 256), 256, 0, stream>>>(x, hB, n);
    }

    dim3 gemm_grid(D_FEAT / BN, (N + BM - 1) / BM);
    int agg_blocks = (N + 7) / 8;     // one wave per node, 8 waves/block

    for (int l = 0; l < L; ++l) {
        const float* W_l = Ws + (size_t)l * D_FEAT * D_FEAT;
        {
            size_t n = (size_t)D_FEAT * D_FEAT;
            gcn_f32_to_bf16<<<(unsigned)((n / 4 + 255) / 256), 256, 0, stream>>>(W_l, Wb, n);
        }
        gcn_gemm_wmma<<<gemm_grid, 256, 0, stream>>>(
            hB, Wb, bs + (size_t)l * D_FEAT, htB, N);
        gcn_agg_ln_relu<<<agg_blocks, 256, 0, stream>>>(
            htB, row_ptr, csr_src, csr_w, inv_deg,
            gammas + (size_t)l * D_FEAT, betas + (size_t)l * D_FEAT,
            out, hB, N);
    }
}
